// GATInfluencePropagation_29368986370259
// MI455X (gfx1250) — matrix operand order repaired and checked by reading, not compiled, not run on hardware
//
#include <hip/hip_runtime.h>
#include <hip/hip_bf16.h>

// ---------------------------------------------------------------------------
// Temporal GAT (GATInfluencePropagation) for MI455X / gfx1250, wave32.
// GEMMs: v_wmma_f32_16x16x32_bf16, 64x128 block tile, software-pipelined
// global->register->LDS staging so VMEM latency hides under the matrix pipe.
// Edge softmax-aggregation: global float atomics. Node epilogues / temporal
// attention: wave32 shuffle butterflies.
// ---------------------------------------------------------------------------

typedef __bf16 bf16;
typedef __attribute__((ext_vector_type(16))) __bf16 v16bf;
typedef __attribute__((ext_vector_type(8)))  __bf16 v8bf;
typedef __attribute__((ext_vector_type(8)))  float  v8f;

#define HIDC 128
#define NEG_SLOPE 0.2f

// ---------------- small helpers ----------------

__device__ __forceinline__ unsigned f2ord(float f) {
  unsigned u = __float_as_uint(f);
  return (u & 0x80000000u) ? ~u : (u | 0x80000000u);
}
__device__ __forceinline__ float ord2f(unsigned u) {
  unsigned b = (u & 0x80000000u) ? (u & 0x7fffffffu) : ~u;
  return __uint_as_float(b);
}
__device__ __forceinline__ float waveAllSum(float x) {
#pragma unroll
  for (int off = 16; off; off >>= 1) x += __shfl_xor(x, off, 32);
  return x;
}

__global__ void fill_f32_k(float* p, float v, long long n) {
  long long i = blockIdx.x * (long long)blockDim.x + threadIdx.x;
  if (i < n) p[i] = v;
}
__global__ void fill_u32_k(unsigned* p, unsigned v, long long n) {
  long long i = blockIdx.x * (long long)blockDim.x + threadIdx.x;
  if (i < n) p[i] = v;
}
__global__ void f32_to_bf16_k(const float* __restrict__ s, bf16* __restrict__ d, long long n) {
  long long i = blockIdx.x * (long long)blockDim.x + threadIdx.x;
  if (i < n) d[i] = (bf16)s[i];
}

// ---------------- WMMA GEMM: C[M,Nc] = A[M,K] * Bt[Nc,K]^T (+bias) ----------
// Block: 256 threads (8 waves). Tile 64x128, K-step 32.
// Wave w: wm = w&1 (32-row half), wn = w>>1 (32-col group); each wave owns a
// 32x32 output patch = 4 accumulators = 4 WMMAs per K step.
// Pipeline: global->regs for slab kb+1 issued before the WMMAs of slab kb.

#define BM 64
#define BN 128
#define BK 32
#define LDP (BK + 8)   // LDS row stride in elements (80B, 16B aligned)

__global__ __launch_bounds__(256) void gemm_bf16_k(
    const float* __restrict__ A, const bf16* __restrict__ Bt,
    const float* __restrict__ bias, float* __restrict__ C,
    int M, int Nc, int K) {
  __shared__ bf16 As[BM * LDP];    //  5,120 B
  __shared__ bf16 Bs[BN * LDP];    // 10,240 B

  const int t    = threadIdx.x;
  const int wave = t >> 5;
  const int lane = t & 31;
  const int hfl  = lane >> 4;      // K-half selector per ISA 16-bit layouts
  const int l16  = lane & 15;
  const int wm   = wave & 1;
  const int wn   = wave >> 1;
  const int mBase = blockIdx.x * BM;
  const int nBase = blockIdx.y * BN;

  // Staging assignment: A: 64 rows x 4 chunks of 8 floats; B: 128 rows x 2
  // chunks of 16 bf16. One thread per chunk.
  const int ar   = t >> 2;
  const int ac8  = (t & 3) * 8;
  const int brr  = t >> 1;
  const int bc16 = (t & 1) * 16;
  const float* Aptr = A + (size_t)(mBase + ar) * K + ac8;
  const bf16*  Bptr = Bt + (size_t)(nBase + brr) * K + bc16;
  const bool   aok  = (mBase + ar) < M;

  float4 a0 = make_float4(0.f, 0.f, 0.f, 0.f), a1 = a0;
  if (aok) { a0 = *(const float4*)Aptr; a1 = *(const float4*)(Aptr + 4); }
  uint4 b0r = *(const uint4*)Bptr;
  uint4 b1r = *(const uint4*)(Bptr + 8);

  v8f acc[4] = {};

  const int nK = K / BK;
  for (int kb = 0; kb < nK; ++kb) {
    __syncthreads();                       // prior WMMAs done reading LDS
    {                                      // commit staged slab to LDS
      v8bf av;
      av[0] = (bf16)a0.x; av[1] = (bf16)a0.y; av[2] = (bf16)a0.z; av[3] = (bf16)a0.w;
      av[4] = (bf16)a1.x; av[5] = (bf16)a1.y; av[6] = (bf16)a1.z; av[7] = (bf16)a1.w;
      *(v8bf*)&As[ar * LDP + ac8]          = av;
      *(uint4*)&Bs[brr * LDP + bc16]       = b0r;
      *(uint4*)&Bs[brr * LDP + bc16 + 8]   = b1r;
    }
    __syncthreads();
    if (kb + 1 < nK) {                     // issue next slab (hidden by WMMAs)
      const float* ap = Aptr + (size_t)(kb + 1) * BK;
      const bf16*  bp = Bptr + (size_t)(kb + 1) * BK;
      a0 = make_float4(0.f, 0.f, 0.f, 0.f); a1 = a0;
      if (aok) { a0 = *(const float4*)ap; a1 = *(const float4*)(ap + 4); }
      b0r = *(const uint4*)bp;
      b1r = *(const uint4*)(bp + 8);
      __builtin_prefetch(ap + BK, 0, 1);   // pull slab kb+2 toward L2
    }

    // Fragments. A 16x32: half0 lanes hold K {0..7,16..23}, half1 {8..15,24..31}.
    // B 32x16: lane = column, half0 K {0..7,16..23}, half1 {8..15,24..31}.
    v16bf afrag[2], bfrag[2];
#pragma unroll
    for (int i = 0; i < 2; ++i) {
      const bf16* arow = &As[(wm * 32 + i * 16 + l16) * LDP];
      v8bf alo = *(const v8bf*)(arow + hfl * 8);
      v8bf ahi = *(const v8bf*)(arow + 16 + hfl * 8);
      const bf16* brow = &Bs[(wn * 32 + i * 16 + l16) * LDP + hfl * 8];
      v8bf blo = *(const v8bf*)brow;
      v8bf bhi = *(const v8bf*)(brow + 16);
#pragma unroll
      for (int j = 0; j < 8; ++j) {
        afrag[i][j] = alo[j]; afrag[i][j + 8] = ahi[j];
        bfrag[i][j] = blo[j]; bfrag[i][j + 8] = bhi[j];
      }
    }
#pragma unroll
    for (int i = 0; i < 2; ++i)
#pragma unroll
      for (int j = 0; j < 2; ++j)
        acc[i * 2 + j] = __builtin_amdgcn_wmma_f32_16x16x32_bf16(
            false, afrag[i], false, bfrag[j], (short)0, acc[i * 2 + j], false, false);
  }

  // C layout: VGPR r -> row (hfl*8 + r), col = lane%16.
#pragma unroll
  for (int i = 0; i < 2; ++i) {
    const int rowb = mBase + wm * 32 + i * 16 + hfl * 8;
#pragma unroll
    for (int j = 0; j < 2; ++j) {
      const int col = nBase + wn * 32 + j * 16 + l16;
      const float bv = bias ? bias[col] : 0.f;
#pragma unroll
      for (int r = 0; r < 8; ++r) {
        const int row = rowb + r;
        if (row < M) C[(size_t)row * Nc + col] = acc[i * 2 + j][r] + bv;
      }
    }
  }
}

// ---------------- attention coefficients: al = sum_c xp[n,h,c]*a[h,c] -------

__global__ __launch_bounds__(256) void att_coef_k(
    const float* __restrict__ xp, const float* __restrict__ aS,
    const float* __restrict__ aD, float* __restrict__ als,
    float* __restrict__ ald, int Nn, int heads) {
  int gw   = (blockIdx.x * blockDim.x + threadIdx.x) >> 5;
  int lane = threadIdx.x & 31;
  if (gw >= Nn * heads) return;
  int n = gw / heads, h = gw % heads;
  const float* xr  = xp + ((size_t)n * heads + h) * HIDC;
  const float* as_ = aS + h * HIDC;
  const float* ad_ = aD + h * HIDC;
  float ss = 0.f, sd = 0.f;
#pragma unroll
  for (int c = lane; c < HIDC; c += 32) {
    float v = xr[c];
    ss += v * as_[c];
    sd += v * ad_[c];
  }
  ss = waveAllSum(ss);
  sd = waveAllSum(sd);
  if (lane == 0) { als[gw] = ss; ald[gw] = sd; }
}

// ---------------- edge passes ----------------

__global__ __launch_bounds__(256) void edge_max_k(
    const int* __restrict__ srcE, const int* __restrict__ dstE, int E_, int Nn,
    const float* __restrict__ als, const float* __restrict__ ald,
    unsigned* __restrict__ mord, int heads) {
  int e = blockIdx.x * blockDim.x + threadIdx.x;
  if (e >= E_ + Nn) return;
  int s = (e < E_) ? srcE[e] : (e - E_);
  int d = (e < E_) ? dstE[e] : (e - E_);
  for (int h = 0; h < heads; ++h) {
    float v = als[s * heads + h] + ald[d * heads + h];
    v = (v > 0.f) ? v : v * NEG_SLOPE;
    atomicMax(&mord[d * heads + h], f2ord(v));
  }
}

__global__ __launch_bounds__(256) void edge_exp_k(
    const int* __restrict__ srcE, const int* __restrict__ dstE, int E_, int Nn,
    const float* __restrict__ als, const float* __restrict__ ald,
    const unsigned* __restrict__ mord, float* __restrict__ sden,
    float* __restrict__ ebuf, int heads) {
  int e = blockIdx.x * blockDim.x + threadIdx.x;
  if (e >= E_ + Nn) return;
  int s = (e < E_) ? srcE[e] : (e - E_);
  int d = (e < E_) ? dstE[e] : (e - E_);
  for (int h = 0; h < heads; ++h) {
    float v = als[s * heads + h] + ald[d * heads + h];
    v = (v > 0.f) ? v : v * NEG_SLOPE;
    float ee = __expf(v - ord2f(mord[d * heads + h]));
    ebuf[(size_t)e * heads + h] = ee;
    atomicAdd(&sden[d * heads + h], ee);
  }
}

// Unnormalized message scatter: acc[dst,c] += e[edge, c/128] * xp[src,c]
__global__ __launch_bounds__(256) void edge_scatter_k(
    const int* __restrict__ srcE, const int* __restrict__ dstE, int E_, int Nn,
    const float* __restrict__ ebuf, const float* __restrict__ xp,
    float* __restrict__ acc, int heads, int F) {
  long long tid = blockIdx.x * (long long)blockDim.x + threadIdx.x;
  const int chunks = F >> 2;
  long long total = (long long)(E_ + Nn) * chunks;
  if (tid >= total) return;
  int e = (int)(tid / chunks);
  int c = (int)(tid % chunks) * 4;
  int s = (e < E_) ? srcE[e] : (e - E_);
  int d = (e < E_) ? dstE[e] : (e - E_);
  int h = c >> 7;  // HID == 128
  float a = ebuf[(size_t)e * heads + h];
  float4 xv = *(const float4*)(xp + (size_t)s * F + c);
  float* ap = acc + (size_t)d * F + c;
  atomicAdd(ap + 0, a * xv.x);
  atomicAdd(ap + 1, a * xv.y);
  atomicAdd(ap + 2, a * xv.z);
  atomicAdd(ap + 3, a * xv.w);
}

// ---------------- per-node epilogue: /sum, +bias, ELU, LayerNorm, residual --
// RES: 0 none, 1 add, 2 pool4(512->128)+add.  One wave per node.

template <int F, int NH, int RES>
__global__ __launch_bounds__(256) void node_epilogue_k(
    const float* __restrict__ acc, const float* __restrict__ sden,
    const float* __restrict__ bias, const float* __restrict__ gamma,
    const float* __restrict__ beta, const float* __restrict__ hres,
    float* __restrict__ outp, int Nn) {
  int gw   = (blockIdx.x * blockDim.x + threadIdx.x) >> 5;
  int lane = threadIdx.x & 31;
  if (gw >= Nn) return;
  constexpr int CNT = F / 32;
  float vals[CNT];
  float mu = 0.f;
#pragma unroll
  for (int i = 0; i < CNT; ++i) {
    int c = lane + 32 * i;
    int h = c >> 7;
    float y = acc[(size_t)gw * F + c] / (sden[gw * NH + h] + 1e-16f) + bias[c];
    y = (y > 0.f) ? y : (__expf(y) - 1.f);   // ELU
    vals[i] = y;
    mu += y;
  }
  mu = waveAllSum(mu) * (1.f / F);
  float var = 0.f;
#pragma unroll
  for (int i = 0; i < CNT; ++i) { float d = vals[i] - mu; var += d * d; }
  var = waveAllSum(var) * (1.f / F);
  float rs = rsqrtf(var + 1e-5f);
#pragma unroll
  for (int i = 0; i < CNT; ++i) {
    int c = lane + 32 * i;
    float z = (vals[i] - mu) * rs * gamma[c] + beta[c];
    if (RES == 1) z += hres[(size_t)gw * F + c];
    if (RES == 2) {
      const float* hr = hres + (size_t)gw * 512 + c * 4;
      z += 0.25f * (hr[0] + hr[1] + hr[2] + hr[3]);
    }
    outp[(size_t)gw * F + c] = z;
  }
}

// ---------------- temporal attention (T=2, 4 heads, hd=32) ------------------

__global__ __launch_bounds__(256) void copy_to_ts_k(
    const float* __restrict__ h, float* __restrict__ ts, int t, int Nn) {
  long long i = blockIdx.x * (long long)blockDim.x + threadIdx.x;
  if (i >= (long long)Nn * HIDC) return;
  long long n = i >> 7;
  int c = (int)(i & 127);
  ts[(size_t)(2 * n + t) * HIDC + c] = h[i];
}

__global__ __launch_bounds__(256) void temporal_attn_k(
    const float* __restrict__ qkv, float* __restrict__ o, int Nn) {
  int gw   = (blockIdx.x * blockDim.x + threadIdx.x) >> 5;
  int lane = threadIdx.x & 31;
  if (gw >= Nn * 4) return;
  int n = gw >> 2, h = gw & 3;
  const float* r0 = qkv + (size_t)(2 * n) * 384;
  const float* r1 = r0 + 384;
  int d = h * 32 + lane;
  float q0 = r0[d], k0 = r0[128 + d], v0 = r0[256 + d];
  float q1 = r1[d], k1 = r1[128 + d], v1 = r1[256 + d];
  const float sc = 0.17677669529663689f;  // 1/sqrt(32)
  float s00 = waveAllSum(q0 * k0) * sc;
  float s01 = waveAllSum(q0 * k1) * sc;
  float s10 = waveAllSum(q1 * k0) * sc;
  float s11 = waveAllSum(q1 * k1) * sc;
  float m0 = fmaxf(s00, s01), m1 = fmaxf(s10, s11);
  float e00 = __expf(s00 - m0), e01 = __expf(s01 - m0);
  float e10 = __expf(s10 - m1), e11 = __expf(s11 - m1);
  float i0 = 1.f / (e00 + e01), i1 = 1.f / (e10 + e11);
  o[(size_t)(2 * n) * HIDC + d]     = (e00 * v0 + e01 * v1) * i0;
  o[(size_t)(2 * n + 1) * HIDC + d] = (e10 * v0 + e11 * v1) * i1;
}

__global__ __launch_bounds__(256) void mean_t_k(
    const float* __restrict__ oo, float* __restrict__ out, int Nn) {
  long long i = blockIdx.x * (long long)blockDim.x + threadIdx.x;
  if (i >= (long long)Nn * HIDC) return;
  long long n = i >> 7;
  int c = (int)(i & 127);
  out[i] = 0.5f * (oo[(size_t)(2 * n) * HIDC + c] + oo[(size_t)(2 * n + 1) * HIDC + c]);
}

// ---------------- host orchestration ----------------

extern "C" void kernel_launch(void* const* d_in, const int* in_sizes, int n_in,
                              void* d_out, int out_size, void* d_ws, size_t ws_size,
                              hipStream_t stream) {
  const float* x    = (const float*)d_in[0];
  const int*   ei   = (const int*)d_in[1];
  const float* Wp   = (const float*)d_in[2];
  const float* bp   = (const float*)d_in[3];
  const float* W0   = (const float*)d_in[4];
  const float* as0  = (const float*)d_in[5];
  const float* ad0  = (const float*)d_in[6];
  const float* b0   = (const float*)d_in[7];
  const float* W1   = (const float*)d_in[8];
  const float* as1  = (const float*)d_in[9];
  const float* ad1  = (const float*)d_in[10];
  const float* b1   = (const float*)d_in[11];
  const float* W2   = (const float*)d_in[12];
  const float* as2  = (const float*)d_in[13];
  const float* ad2  = (const float*)d_in[14];
  const float* b2   = (const float*)d_in[15];
  const float* g0   = (const float*)d_in[16];
  const float* be0  = (const float*)d_in[17];
  const float* g1   = (const float*)d_in[18];
  const float* be1  = (const float*)d_in[19];
  const float* g2   = (const float*)d_in[20];
  const float* be2  = (const float*)d_in[21];
  const float* Wqkv = (const float*)d_in[22];
  const float* bqkv = (const float*)d_in[23];
  const float* Wo   = (const float*)d_in[24];
  const float* bo   = (const float*)d_in[25];

  const int N  = in_sizes[0] / 128;   // 20000
  const int E  = in_sizes[1] / 2;     // 256000
  const int Et = E + N;               // with self loops
  const int* srcE = ei;
  const int* dstE = ei + E;

  // workspace layout
  char* ws = (char*)d_ws;
  size_t off = 0;
  auto alloc = [&](size_t bytes) -> char* {
    char* p = ws + off;
    off += (bytes + 255) & ~(size_t)255;
    return p;
  };
  bf16* WpB = (bf16*)alloc(16384 * 2);
  bf16* W0B = (bf16*)alloc(65536 * 2);
  bf16* W1B = (bf16*)alloc(262144 * 2);
  bf16* W2B = (bf16*)alloc(65536 * 2);
  bf16* WqB = (bf16*)alloc(49152 * 2);
  bf16* WoB = (bf16*)alloc(16384 * 2);
  float* hbuf0 = (float*)alloc((size_t)N * 512 * 4);
  float* hbuf1 = (float*)alloc((size_t)N * 512 * 4);
  float* xp    = (float*)alloc((size_t)N * 512 * 4);   // qkv aliases xp+acc
  float* acc   = (float*)alloc((size_t)N * 512 * 4);
  float* als   = (float*)alloc((size_t)N * 4 * 4);
  float* ald   = (float*)alloc((size_t)N * 4 * 4);
  unsigned* mord = (unsigned*)alloc((size_t)N * 4 * 4);
  float* sden  = (float*)alloc((size_t)N * 4 * 4);
  float* ebuf  = (float*)alloc((size_t)Et * 4 * 4);
  float* ts    = (float*)alloc((size_t)N * 2 * 128 * 4);
  float* qkvb  = xp;      // [2N, 384] fits in xp+acc (contiguous)
  float* obuf  = hbuf0;   // reused after GAT timesteps
  float* oobuf = hbuf1;
  (void)off; (void)ws_size; (void)n_in; (void)out_size;

  auto cgrid = [](long long n) { return dim3((unsigned)((n + 255) / 256)); };

  // weight conversion to bf16 (once per launch; deterministic)
  f32_to_bf16_k<<<cgrid(16384), 256, 0, stream>>>(Wp, WpB, 16384);
  f32_to_bf16_k<<<cgrid(65536), 256, 0, stream>>>(W0, W0B, 65536);
  f32_to_bf16_k<<<cgrid(262144), 256, 0, stream>>>(W1, W1B, 262144);
  f32_to_bf16_k<<<cgrid(65536), 256, 0, stream>>>(W2, W2B, 65536);
  f32_to_bf16_k<<<cgrid(49152), 256, 0, stream>>>(Wqkv, WqB, 49152);
  f32_to_bf16_k<<<cgrid(16384), 256, 0, stream>>>(Wo, WoB, 16384);

  auto gemm = [&](const float* A, const bf16* B, const float* bias, float* C,
                  int M, int Nc, int K) {
    dim3 g((M + BM - 1) / BM, Nc / BN);
    gemm_bf16_k<<<g, 256, 0, stream>>>(A, B, bias, C, M, Nc, K);
  };

  auto gat = [&](const float* hin, int Fin, const bf16* WB, const float* aS,
                 const float* aD, const float* bias, const float* gamma,
                 const float* beta, const float* hres, int resmode, float* hout,
                 int heads) {
    const int F = heads * HIDC;
    gemm(hin, WB, nullptr, xp, N, F, Fin);
    att_coef_k<<<cgrid((long long)N * heads * 32), 256, 0, stream>>>(xp, aS, aD, als, ald, N, heads);
    fill_u32_k<<<cgrid((long long)N * heads), 256, 0, stream>>>(mord, 0u, (long long)N * heads);
    fill_f32_k<<<cgrid((long long)N * heads), 256, 0, stream>>>(sden, 0.f, (long long)N * heads);
    fill_f32_k<<<cgrid((long long)N * F), 256, 0, stream>>>(acc, 0.f, (long long)N * F);
    edge_max_k<<<cgrid(Et), 256, 0, stream>>>(srcE, dstE, E, N, als, ald, mord, heads);
    edge_exp_k<<<cgrid(Et), 256, 0, stream>>>(srcE, dstE, E, N, als, ald, mord, sden, ebuf, heads);
    edge_scatter_k<<<cgrid((long long)Et * (F / 4)), 256, 0, stream>>>(srcE, dstE, E, N, ebuf, xp, acc, heads, F);
    dim3 eg((unsigned)(((long long)N * 32 + 255) / 256));
    if (heads == 4 && resmode == 0)
      node_epilogue_k<512, 4, 0><<<eg, 256, 0, stream>>>(acc, sden, bias, gamma, beta, hres, hout, N);
    else if (heads == 4 && resmode == 1)
      node_epilogue_k<512, 4, 1><<<eg, 256, 0, stream>>>(acc, sden, bias, gamma, beta, hres, hout, N);
    else
      node_epilogue_k<128, 1, 2><<<eg, 256, 0, stream>>>(acc, sden, bias, gamma, beta, hres, hout, N);
  };

  // input projection: h = x @ Wp^T + bp  -> hbuf0 [N,128]
  gemm(x, WpB, bp, hbuf0, N, 128, 128);

  float* X = hbuf0;
  float* Y = hbuf1;
  for (int t = 0; t < 2; ++t) {
    gat(X, 128, W0B, as0, ad0, b0, g0, be0, nullptr, 0, Y, 4);  // L0: X(128)->Y(512)
    gat(Y, 512, W1B, as1, ad1, b1, g1, be1, Y, 1, X, 4);        // L1: Y->X, res=Y
    gat(X, 512, W2B, as2, ad2, b2, g2, be2, X, 2, Y, 1);        // L2: X->Y(128), res=pool4(X)
    copy_to_ts_k<<<cgrid((long long)N * HIDC), 256, 0, stream>>>(Y, ts, t, N);
    float* tmp = X; X = Y; Y = tmp;  // next timestep input = this state
  }

  // temporal self-attention over T=2
  gemm(ts, WqB, bqkv, qkvb, 2 * N, 384, 128);                       // qkv
  temporal_attn_k<<<cgrid((long long)N * 4 * 32), 256, 0, stream>>>(qkvb, obuf, N);
  gemm(obuf, WoB, bo, oobuf, 2 * N, 128, 128);                      // out proj
  mean_t_k<<<cgrid((long long)N * HIDC), 256, 0, stream>>>(oobuf, (float*)d_out, N);
}
